// FreMaskBlock_35210141893343
// MI455X (gfx1250) — matrix operand order repaired and checked by reading, not compiled, not run on hardware
//
#include <hip/hip_runtime.h>
#include <hip/hip_bf16.h>
#include <math.h>

// ---------------- CDNA5 WMMA types & helpers ----------------
typedef __attribute__((ext_vector_type(16))) __bf16 v16bf;
typedef __attribute__((ext_vector_type(8)))  __bf16 v8bf;
typedef __attribute__((ext_vector_type(8)))  float  v8f;

__device__ __forceinline__ v16bf bf_cat(v8bf lo, v8bf hi) {
  return __builtin_shufflevector(lo, hi, 0,1,2,3,4,5,6,7,8,9,10,11,12,13,14,15);
}

// A-matrix 16x32 bf16 tile (M=16,K=32), row-major source with leading dim ldk.
// Per ISA layout: lanes 0-15 -> row m0+lane, VGPR0-3 = K[0..7], VGPR4-7 = K[16..23];
// lanes 16-31 -> same rows, K[8..15] and K[24..31].
__device__ __forceinline__ v16bf load_a16x32(const __bf16* base, int ldk,
                                             int m0, int k0, int l16, int lh) {
  const __bf16* p = base + (size_t)(m0 + l16) * (size_t)ldk + k0 + lh * 8;
  v8bf lo = *(const v8bf*)p;         // K[k0 + 8*lh .. +7]
  v8bf hi = *(const v8bf*)(p + 16);  // K[k0 + 16 + 8*lh .. +7]
  return bf_cat(lo, hi);
}

// B-matrix 32x16 bf16 tile (K=32,N=16) from LDS stored [n][k] (k contiguous,
// row stride `stride` elements). Lanes 0-15: col n0+lane, K[k0..k0+7] and K[k0+16..k0+23];
// lanes 16-31: K[k0+8..k0+15] and K[k0+24..k0+31].
__device__ __forceinline__ v16bf load_b32x16(const __bf16* lds, int stride,
                                             int n0, int k0, int l16, int lh) {
  const __bf16* p = lds + (size_t)(n0 + l16) * (size_t)stride + k0 + lh * 8;
  v8bf lo = *(const v8bf*)p;
  v8bf hi = *(const v8bf*)(p + 16);
  return bf_cat(lo, hi);
}

__device__ __forceinline__ v8f wmma_bf16(v16bf a, v16bf b, v8f c) {
  // D = A*B + C : v_wmma_f32_16x16x32_bf16
  return __builtin_amdgcn_wmma_f32_16x16x32_bf16(false, a, false, b,
                                                 (short)0, c, false, false);
}

__device__ __forceinline__ float gelu_exact(float x) {
  return 0.5f * x * (1.0f + erff(x * 0.70710678118654752440f));
}

__device__ __forceinline__ v8bf to_bf8(v8f v) {
  v8bf r;
#pragma unroll
  for (int i = 0; i < 8; ++i) r[i] = (__bf16)v[i];
  return r;
}

// ---------------- constants ----------------
#define BATCH 16
#define CH    256
#define HW    16384          // 128*128
#define ROWPIX 128           // pixels per workgroup row tile
#define XN_STRIDE 264        // 256 + 8 pad (bf16 elems)
#define Y_STRIDE  136        // 128 + 8 pad
#define Y2_STRIDE 72         // 64 + 8 pad

// ---------------- kernel 0: weights fp32 -> bf16 ----------------
__global__ __launch_bounds__(256) void kPrep(const float* __restrict__ w0,
                                             const float* __restrict__ w1,
                                             const float* __restrict__ w2,
                                             __bf16* __restrict__ W0bf,
                                             __bf16* __restrict__ W1bf,
                                             __bf16* __restrict__ W2bf) {
  int i = blockIdx.x * 256 + threadIdx.x;
  if (i < 65536) {
    W0bf[i] = (__bf16)w0[i];                       // 256x256
  } else if (i < 81920) {
    int j = i - 65536;                             // W1 local half: [128][0..127]
    W1bf[j] = (__bf16)w1[(j >> 7) * 256 + (j & 127)];
  } else if (i < 90112) {
    int j = i - 81920;                             // 64x128
    W2bf[j] = (__bf16)w2[j];
  }
}

// ---------------- kernel A: add + LayerNorm + conv0(GEMM,WMMA) + GELU ------
// grid = 2048 (b*128 rows), block = 256 (8 waves)
__global__ __launch_bounds__(256) void kA(const float* __restrict__ dec,
                                          const float* __restrict__ enc,
                                          const float* __restrict__ ln_w,
                                          const float* __restrict__ ln_b,
                                          const __bf16* __restrict__ W0bf,
                                          const float* __restrict__ b0,
                                          __bf16* __restrict__ y0l,
                                          float* __restrict__ pool) {
  extern __shared__ char smem_raw[];
  __bf16* xn   = (__bf16*)smem_raw;                    // [128][264] bf16 = 67584 B
  float*  s0   = (float*)(smem_raw + 67584);           // [2][128]
  float*  s1   = s0 + 256;                             // [2][128]
  float*  muA  = s1 + 256;                             // [128]
  float*  rsA  = muA + 128;                            // [128]
  float*  lw   = rsA + 128;                            // [256]
  float*  lb   = lw + 256;                             // [256]

  const int t   = threadIdx.x;
  const int bb  = blockIdx.x >> 7;
  const int row = blockIdx.x & 127;

  lw[t] = ln_w[t];
  lb[t] = ln_b[t];

  // ---- pass 1: x = dec + enc, accumulate moments, store raw bf16 tile ----
  const int p  = t & 127;        // pixel in row
  const int chv = t >> 7;        // which channel interleave (0/1)
  const size_t gbase = (size_t)bb * CH * HW + (size_t)row * ROWPIX + p;
  float sum = 0.f, ssq = 0.f;
#pragma unroll 4
  for (int j = 0; j < 64; ++j) {
    int c = chv * 2 + j * 4;     // thread covers channels {c, c+1}
    size_t i0 = gbase + (size_t)c * HW;
    size_t i1 = i0 + HW;
    float x0 = dec[i0] + enc[i0];
    float x1 = dec[i1] + enc[i1];
    sum += x0 + x1;
    ssq += x0 * x0 + x1 * x1;
    __bf16* d = xn + p * XN_STRIDE + c;
    d[0] = (__bf16)x0;
    d[1] = (__bf16)x1;
  }
  s0[chv * 128 + p] = sum;
  s1[chv * 128 + p] = ssq;
  __syncthreads();

  // ---- pass 2: per-pixel mean / rstd ----
  if (t < 128) {
    float s = s0[t] + s0[128 + t];
    float q = s1[t] + s1[128 + t];
    float m = s * (1.0f / 256.0f);
    float v = q * (1.0f / 256.0f) - m * m;
    muA[t] = m;
    rsA[t] = rsqrtf(v + 1e-6f);
  }
  __syncthreads();

  // ---- pass 3: normalize in place ----
  {
    float m = muA[p], r = rsA[p];
#pragma unroll 4
    for (int j = 0; j < 64; ++j) {
      int c = chv * 2 + j * 4;
      __bf16* d = xn + p * XN_STRIDE + c;
      float x0 = (float)d[0], x1 = (float)d[1];
      d[0] = (__bf16)((x0 - m) * r * lw[c]     + lb[c]);
      d[1] = (__bf16)((x1 - m) * r * lw[c + 1] + lb[c + 1]);
    }
  }
  __syncthreads();

  // ---- GEMM0: y0[o,p] = W0[o,:] . xn[:,p] + b0[o], M=256 N=128 K=256 ----
  const int wv  = t >> 5;
  const int ln  = t & 31;
  const int l16 = ln & 15;
  const int lh  = ln >> 4;
  const int m0a = wv * 32;
  const int m0b = m0a + 16;

  v8f acc_a[8], acc_b[8];
  v8f ba = *(const v8f*)(b0 + m0a + 8 * lh);
  v8f bc = *(const v8f*)(b0 + m0b + 8 * lh);
#pragma unroll
  for (int nt = 0; nt < 8; ++nt) { acc_a[nt] = ba; acc_b[nt] = bc; }

#pragma unroll
  for (int k0 = 0; k0 < 256; k0 += 32) {
    v16bf Aa = load_a16x32(W0bf, 256, m0a, k0, l16, lh);
    v16bf Ab = load_a16x32(W0bf, 256, m0b, k0, l16, lh);
#pragma unroll
    for (int nt = 0; nt < 8; ++nt) {
      v16bf B = load_b32x16(xn, XN_STRIDE, nt * 16, k0, l16, lh);
      acc_a[nt] = wmma_bf16(Aa, B, acc_a[nt]);
      acc_b[nt] = wmma_bf16(Ab, B, acc_b[nt]);
    }
  }

  const size_t pixbase = (size_t)bb * HW + (size_t)row * ROWPIX;
  if (wv < 4) {
    // channels 0..127 : gelu, store bf16 [pixel][channel]
#pragma unroll
    for (int nt = 0; nt < 8; ++nt) {
      int pix = nt * 16 + l16;
      v8f ga, gb;
#pragma unroll
      for (int i = 0; i < 8; ++i) { ga[i] = gelu_exact(acc_a[nt][i]); gb[i] = gelu_exact(acc_b[nt][i]); }
      *(v8bf*)(y0l + (pixbase + pix) * 128 + m0a + 8 * lh) = to_bf8(ga);
      *(v8bf*)(y0l + (pixbase + pix) * 128 + m0b + 8 * lh) = to_bf8(gb);
    }
  } else {
    // channels 128..255 : gelu, reduce over the 128 pixels -> per-WG partials
    v8f pa, pb;
#pragma unroll
    for (int i = 0; i < 8; ++i) { pa[i] = 0.f; pb[i] = 0.f; }
#pragma unroll
    for (int nt = 0; nt < 8; ++nt) {
#pragma unroll
      for (int i = 0; i < 8; ++i) {
        pa[i] += gelu_exact(acc_a[nt][i]);
        pb[i] += gelu_exact(acc_b[nt][i]);
      }
    }
#pragma unroll
    for (int off = 1; off < 16; off <<= 1) {
#pragma unroll
      for (int i = 0; i < 8; ++i) {
        pa[i] += __shfl_xor(pa[i], off, 32);
        pb[i] += __shfl_xor(pb[i], off, 32);
      }
    }
    if (l16 == 0) {  // lanes 0 and 16 hold the totals for their 8 rows
      float* dst = pool + (size_t)blockIdx.x * 128;
#pragma unroll
      for (int i = 0; i < 8; ++i) {
        dst[(m0a - 128) + 8 * lh + i] = pa[i];
        dst[(m0b - 128) + 8 * lh + i] = pb[i];
      }
    }
  }
}

// ---------------- kernel B: pool reduce + fused conv1 bias ----------------
// grid = 16 (batches), block = 128
__global__ __launch_bounds__(128) void kB(const float* __restrict__ pool,
                                          const float* __restrict__ w1,
                                          const float* __restrict__ b1,
                                          float* __restrict__ bias1) {
  __shared__ float g[128];
  const int bb = blockIdx.x, t = threadIdx.x;
  float s = 0.f;
  for (int r = 0; r < 128; ++r)
    s += pool[((size_t)bb * 128 + r) * 128 + t];
  g[t] = s * (1.0f / 16384.0f);   // mean over H*W
  __syncthreads();
  float acc = b1[t];
  for (int k = 0; k < 128; ++k)
    acc += w1[t * 256 + 128 + k] * g[k];   // global half of W1
  bias1[bb * 128 + t] = acc;
}

// ---------------- kernel C: conv1 -> conv2 -> conv3 -> sigmoid (WMMA) -----
// grid = 2048, block = 256
__global__ __launch_bounds__(256) void kC(const __bf16* __restrict__ y0l,
                                          const __bf16* __restrict__ W1bf,
                                          const float* __restrict__ bias1,
                                          const __bf16* __restrict__ W2bf,
                                          const float* __restrict__ b2,
                                          const float* __restrict__ w3,
                                          const float* __restrict__ b3,
                                          float* __restrict__ out) {
  extern __shared__ char smem_raw[];
  __bf16* y0s = (__bf16*)smem_raw;                 // [128][136] = 34816 B
  __bf16* y1s = (__bf16*)(smem_raw + 34816);       // [128][136] = 34816 B
  __bf16* y2s = (__bf16*)(smem_raw + 69632);       // [128][72]  = 18432 B
  float*  w3s = (float*)(smem_raw + 88064);        // [64]

  const int t   = threadIdx.x;
  const int bb  = blockIdx.x >> 7;
  const int row = blockIdx.x & 127;
  const size_t pixbase = (size_t)bb * HW + (size_t)row * ROWPIX;

  // stage y0-local tile (pixels 0..255 x ch 0..63, bf16) into LDS, [p][c]
  for (int q = t; q < 2048; q += 256) {
    int pp = q >> 3, c8 = (q & 7) * 8;
    *(v8bf*)(y0s + pp * Y_STRIDE + c8) =
        *(const v8bf*)(y0l + (pixbase + pp) * 128 + c8);
  }
  if (t < 64) w3s[t] = w3[t];
  __syncthreads();

  const int wv  = t >> 5;
  const int ln  = t & 31;
  const int l16 = ln & 15;
  const int lh  = ln >> 4;

  // ---- GEMM1: M=128 N=128 K=128, wave wv owns M rows [wv*16, wv*16+16) ----
  {
    const int m0 = wv * 16;
    v8f acc[8];
    v8f bias = *(const v8f*)(bias1 + bb * 128 + m0 + 8 * lh);
#pragma unroll
    for (int nt = 0; nt < 8; ++nt) acc[nt] = bias;
#pragma unroll
    for (int k0 = 0; k0 < 128; k0 += 32) {
      v16bf A = load_a16x32(W1bf, 128, m0, k0, l16, lh);
#pragma unroll
      for (int nt = 0; nt < 8; ++nt) {
        v16bf B = load_b32x16(y0s, Y_STRIDE, nt * 16, k0, l16, lh);
        acc[nt] = wmma_bf16(A, B, acc[nt]);
      }
    }
#pragma unroll
    for (int nt = 0; nt < 8; ++nt) {
      int pp = nt * 16 + l16;
      v8f g;
#pragma unroll
      for (int i = 0; i < 8; ++i) g[i] = gelu_exact(acc[nt][i]);
      *(v8bf*)(y1s + pp * Y_STRIDE + m0 + 8 * lh) = to_bf8(g);
    }
  }
  __syncthreads();

  // ---- GEMM2: M=64 N=128 K=128; wave wv: m0=(wv&3)*16, N tiles (wv>>2)*4.. ---
  {
    const int m0 = (wv & 3) * 16;
    const int nb = (wv >> 2) * 4;
    v8f acc[4];
    v8f bias = *(const v8f*)(b2 + m0 + 8 * lh);
#pragma unroll
    for (int j = 0; j < 4; ++j) acc[j] = bias;
#pragma unroll
    for (int k0 = 0; k0 < 128; k0 += 32) {
      v16bf A = load_a16x32(W2bf, 128, m0, k0, l16, lh);
#pragma unroll
      for (int j = 0; j < 4; ++j) {
        v16bf B = load_b32x16(y1s, Y_STRIDE, (nb + j) * 16, k0, l16, lh);
        acc[j] = wmma_bf16(A, B, acc[j]);
      }
    }
#pragma unroll
    for (int j = 0; j < 4; ++j) {
      int pp = (nb + j) * 16 + l16;
      v8f g;
#pragma unroll
      for (int i = 0; i < 8; ++i) g[i] = gelu_exact(acc[j][i]);
      *(v8bf*)(y2s + pp * Y2_STRIDE + m0 + 8 * lh) = to_bf8(g);
    }
  }
  __syncthreads();

  // ---- conv3 (64 -> 1) + sigmoid ----
  if (t < 128) {
    float acc = b3[0];
#pragma unroll 8
    for (int k = 0; k < 64; ++k)
      acc += (float)y2s[t * Y2_STRIDE + k] * w3s[k];
    out[pixbase + t] = 1.0f / (1.0f + expf(-acc));
  }
}

// ---------------- launcher ----------------
extern "C" void kernel_launch(void* const* d_in, const int* in_sizes, int n_in,
                              void* d_out, int out_size, void* d_ws, size_t ws_size,
                              hipStream_t stream) {
  (void)in_sizes; (void)n_in; (void)out_size; (void)ws_size;

  const float* dec  = (const float*)d_in[0];
  const float* enc  = (const float*)d_in[1];
  const float* ln_w = (const float*)d_in[2];
  const float* ln_b = (const float*)d_in[3];
  const float* w0   = (const float*)d_in[4];
  const float* b0   = (const float*)d_in[5];
  const float* w1   = (const float*)d_in[6];
  const float* b1   = (const float*)d_in[7];
  const float* w2   = (const float*)d_in[8];
  const float* b2   = (const float*)d_in[9];
  const float* w3   = (const float*)d_in[10];
  const float* b3   = (const float*)d_in[11];
  float* out = (float*)d_out;

  char* ws = (char*)d_ws;
  constexpr size_t OFF_Y0    = 0;                          // 64 MiB bf16 y0-local
  constexpr size_t OFF_W0    = 67108864;                   // 128 KiB
  constexpr size_t OFF_W1    = OFF_W0 + 131072;            // 32 KiB
  constexpr size_t OFF_W2    = OFF_W1 + 32768;             // 16 KiB
  constexpr size_t OFF_POOL  = OFF_W2 + 16384;             // 1 MiB (2048*128 f32)
  constexpr size_t OFF_BIAS1 = OFF_POOL + 2048 * 128 * 4;  // 8 KiB

  __bf16* W0bf  = (__bf16*)(ws + OFF_W0);
  __bf16* W1bf  = (__bf16*)(ws + OFF_W1);
  __bf16* W2bf  = (__bf16*)(ws + OFF_W2);
  __bf16* y0l   = (__bf16*)(ws + OFF_Y0);
  float*  pool  = (float*)(ws + OFF_POOL);
  float*  bias1 = (float*)(ws + OFF_BIAS1);

  kPrep<<<352, 256, 0, stream>>>(w0, w1, w2, W0bf, W1bf, W2bf);
  kA<<<2048, 256, 72704, stream>>>(dec, enc, ln_w, ln_b, W0bf, b0, y0l, pool);
  kB<<<16, 128, 0, stream>>>(pool, w1, b1, bias1);
  kC<<<2048, 256, 88320, stream>>>(y0l, W1bf, bias1, W2bf, b2, w3, b3, out);
}
